// Decoder_51780125721064
// MI455X (gfx1250) — matrix operand order repaired and checked by reading, not compiled, not run on hardware
//
#include <hip/hip_runtime.h>
#include <stdint.h>

// ---------------------------------------------------------------------------
// Types for CDNA5 WMMA / TDM
// ---------------------------------------------------------------------------
typedef __attribute__((ext_vector_type(16))) _Float16 v16h;
typedef __attribute__((ext_vector_type(8)))  _Float16 v8h;
typedef __attribute__((ext_vector_type(8)))  float    v8f;
typedef __attribute__((ext_vector_type(4)))  unsigned int v4u;
typedef __attribute__((ext_vector_type(8)))  int      v8i;
typedef __attribute__((ext_vector_type(4)))  int      v4i;

#define DEC_T   500
#define BATCH   16
#define ENC_TT  256
#define ENC_DD  512
#define FEATD   1025
#define NBLK    32          // persistent grid: 32 blocks x 256 threads = 256 waves

// ---------------------------------------------------------------------------
// Small device helpers
// ---------------------------------------------------------------------------
__device__ __forceinline__ float rng01(unsigned seed, unsigned i) {
    unsigned x = seed ^ (i * 0x9E3779B9u);
    x ^= x >> 16; x *= 0x7FEB352Du;
    x ^= x >> 15; x *= 0x846CA68Bu;
    x ^= x >> 16;
    return (float)(x >> 8) * (1.0f / 16777216.0f);
}
__device__ __forceinline__ float sigm(float x) { return 1.0f / (1.0f + __expf(-x)); }

// Device-wide sense-reversal barrier (persistent kernel).  bar[0]=count, bar[1]=gen.
__device__ __forceinline__ void grid_sync(unsigned* bar) {
    __syncthreads();
    if (threadIdx.x == 0) {
        __threadfence();
        unsigned gen = __atomic_load_n(bar + 1, __ATOMIC_RELAXED);
        unsigned arrived = atomicAdd(bar, 1u) + 1u;
        if (arrived == (unsigned)NBLK) {
            __atomic_store_n(bar, 0u, __ATOMIC_RELAXED);
            __threadfence();
            atomicAdd(bar + 1, 1u);
        } else {
            while (__atomic_load_n(bar + 1, __ATOMIC_RELAXED) == gen) {
                __builtin_amdgcn_s_sleep(2);
            }
        }
        __threadfence();
    }
    __syncthreads();
}

// ---------------------------------------------------------------------------
// Tensor Data Mover: 1D DMA of nElem 2-byte elements global -> LDS.
// Descriptor per CDNA5 ISA 8.3/8.4: group0 = {count=1, lds_addr, global_addr,
// type=2}; group1 = {data_size=1(2B), tensor_dim0=tile_dim0=nElem,
// dim0_stride=nElem}; groups 2/3 zero (<=2D tensor).  Issue from ONE wave
// (TDM ignores EXEC; one issue per wave), then s_wait_tensorcnt.
// ---------------------------------------------------------------------------
__device__ __forceinline__ void tdm_load_1d(const void* gsrc, void* ldst, unsigned nElem) {
    unsigned long long ga = (unsigned long long)(uintptr_t)gsrc;
    unsigned lofs = (unsigned)(uintptr_t)ldst;   // LDS aperture: addr[31:0] = LDS offset
    v4u g0;
    g0[0] = 1u;                                          // count=1 (valid user D#)
    g0[1] = lofs;                                        // lds_addr [63:32]
    g0[2] = (unsigned)(ga & 0xFFFFFFFFu);                // global_addr [95:64]
    g0[3] = (unsigned)((ga >> 32) & 0x01FFFFFFu) | (2u << 30);   // addr[56:32] | type=2
    v8i g1;
    g1[0] = 0x00010000;                                  // data_size=1 -> 2 bytes
    g1[1] = (int)((nElem & 0xFFFFu) << 16);              // tensor_dim0[15:0]
    g1[2] = (int)((nElem >> 16) & 0xFFFFu);              // tensor_dim0[31:16], tensor_dim1=0
    g1[3] = (int)((nElem & 0xFFFFu) << 16);              // tile_dim0 = nElem
    g1[4] = 0;                                           // tile_dim1/2 unused
    g1[5] = (int)nElem;                                  // tensor_dim0_stride[31:0]
    g1[6] = 0;
    g1[7] = 0;
    v4i z; z[0] = 0; z[1] = 0; z[2] = 0; z[3] = 0;
#if defined(__clang_major__) && (__clang_major__ >= 23)
    v8i z8; for (int i = 0; i < 8; ++i) z8[i] = 0;
    __builtin_amdgcn_tensor_load_to_lds(g0, g1, z, z, z8, 0);
#else
    __builtin_amdgcn_tensor_load_to_lds(g0, g1, z, z, 0);
#endif
    __builtin_amdgcn_s_wait_tensorcnt(0);
}

// ---------------------------------------------------------------------------
// Wave-level GEMM tile: D(16x16,f32) = A(16 x 32*nchunks, f16) * Bpacked
// ---------------------------------------------------------------------------
__device__ __forceinline__ v8f wmma_tile(const _Float16* A, int lda,
                                         const _Float16* Bp, int nchunks, int lane) {
    const int row = lane & 15, half = lane >> 4;
    v8f c;
#pragma unroll
    for (int i = 0; i < 8; ++i) c[i] = 0.0f;
    const _Float16* arow = A + (size_t)row * lda + half * 8;
    const _Float16* bp   = Bp + (size_t)lane * 16;
    for (int kc = 0; kc < nchunks; ++kc) {
        __builtin_prefetch(bp + (size_t)(kc + 4) * 512, 0, 1);  // global_prefetch_b8
        v8h lo = *(const v8h*)(arow + kc * 32);
        v8h hi = *(const v8h*)(arow + kc * 32 + 16);
        v16h a = __builtin_shufflevector(lo, hi, 0,1,2,3,4,5,6,7,8,9,10,11,12,13,14,15);
        v16h b = *(const v16h*)(bp + (size_t)kc * 512);
        c = __builtin_amdgcn_wmma_f32_16x16x32_f16(false, a, false, b,
                                                   (short)0, c, false, false);
    }
    return c;
}

// ---------------------------------------------------------------------------
// Weight packing: W (K x N row-major f32) -> f16 WMMA B-fragments.
// lane L holds  W[kc*32 + (j>>3)*16 + (L>>4)*8 + (j&7) , n*16 + (L&15)]  for j=0..15.
// ---------------------------------------------------------------------------
__global__ void pack_weights(const float* __restrict__ W, _Float16* __restrict__ dst,
                             int K, int N, int KCsrc, int chunkOff, int KCtot) {
    int blk  = blockIdx.x;
    int tile = blk / KCsrc;
    int kc   = blk % KCsrc;
    int lane = threadIdx.x;
    int col  = tile * 16 + (lane & 15);
    int half = lane >> 4;
    _Float16* out = dst + (((size_t)tile * KCtot + chunkOff + kc) * 32 + lane) * 16;
#pragma unroll
    for (int j = 0; j < 16; ++j) {
        int k = kc * 32 + (j >> 3) * 16 + half * 8 + (j & 7);
        float v = (k < K && col < N) ? W[(size_t)k * N + col] : 0.0f;
        out[j] = (_Float16)v;
    }
}

// Teacher-forced decoder inputs -> f16  [t][b][1056] (zero-padded past 1025)
__global__ void build_feats(const float* __restrict__ features,
                            const float* __restrict__ go, _Float16* __restrict__ out) {
    size_t total = (size_t)DEC_T * BATCH * 1056;
    for (size_t i = (size_t)blockIdx.x * blockDim.x + threadIdx.x; i < total;
         i += (size_t)gridDim.x * blockDim.x) {
        int f = (int)(i % 1056);
        size_t r = i / 1056;
        int b = (int)(r % BATCH);
        int t = (int)(r / BATCH);
        float v = 0.0f;
        if (f < FEATD) v = (t == 0) ? go[f] : features[((size_t)b * FEATD + f) * DEC_T + (t - 1)];
        out[i] = (_Float16)v;
    }
}

// Prenet for all 500 tiles of 16 rows: two f16 WMMA layers, LDS staging between.
__global__ __launch_bounds__(256)
void prenet_kernel(const _Float16* __restrict__ feats16,
                   const _Float16* __restrict__ pw0, const _Float16* __restrict__ pw1,
                   _Float16* __restrict__ xf16) {
    __shared__ __align__(32) _Float16 s_h0[16 * 256];
    int t = blockIdx.x;
    int wave = threadIdx.x >> 5, lane = threadIdx.x & 31;
    const _Float16* A0 = feats16 + (size_t)t * 16 * 1056;
    for (int tile = wave; tile < 16; tile += 8) {
        v8f c = wmma_tile(A0, 1056, pw0 + (size_t)tile * 33 * 512, 33, lane);
        int col = tile * 16 + (lane & 15), half = lane >> 4;
#pragma unroll
        for (int r = 0; r < 8; ++r) {
            int b = r + half * 8;
            float h = fmaxf(c[r], 0.0f);
            h = (rng01(0x1111u, (unsigned)((t * 16 + b) * 256 + col)) < 0.5f) ? h * 2.0f : 0.0f;
            s_h0[b * 256 + col] = (_Float16)h;
        }
    }
    __syncthreads();
    for (int tile = wave; tile < 16; tile += 8) {
        v8f c = wmma_tile(s_h0, 256, pw1 + (size_t)tile * 8 * 512, 8, lane);
        int col = tile * 16 + (lane & 15), half = lane >> 4;
#pragma unroll
        for (int r = 0; r < 8; ++r) {
            int b = r + half * 8;
            float h = fmaxf(c[r], 0.0f);
            h = (rng01(0x2222u, (unsigned)((t * 16 + b) * 256 + col)) < 0.5f) ? h * 2.0f : 0.0f;
            xf16[((size_t)t * 16 + b) * 256 + col] = (_Float16)h;
        }
    }
}

// proc_mem[b][tt][128] = mem[b,tt,:] @ attn_memory_w   (one-time)
__global__ void proc_mem_kernel(const float* __restrict__ enc,
                                const float* __restrict__ wm, float* __restrict__ pm) {
    int bid = blockIdx.x;                 // BATCH*ENC_TT
    int b = bid >> 8, tt = bid & 255;
    int j = threadIdx.x;                  // 128
    float acc = 0.0f;
    for (int d = 0; d < ENC_DD; ++d)
        acc += enc[((size_t)b * ENC_DD + d) * ENC_TT + tt] * wm[d * 128 + j];
    pm[((size_t)b * ENC_TT + tt) * 128 + j] = acc;
}

__global__ void init_state(float* preh, float* prec, float* postc, float* ctx,
                           float* algn, float* cum,
                           _Float16* acat, _Float16* pacat, _Float16* pjacat,
                           unsigned* bar) {
    size_t i0 = (size_t)blockIdx.x * blockDim.x + threadIdx.x;
    size_t st = (size_t)gridDim.x * blockDim.x;
    if (i0 == 0) { bar[0] = 0u; bar[1] = 0u; }
    for (size_t i = i0; i < (size_t)BATCH * 1024; i += st) { preh[i] = 0.f; prec[i] = 0.f; postc[i] = 0.f; }
    for (size_t i = i0; i < (size_t)BATCH * ENC_DD; i += st) ctx[i] = 0.f;
    for (size_t i = i0; i < (size_t)BATCH * ENC_TT; i += st) {
        float v = ((i & 255) == 0) ? 1.0f : 0.0f;
        algn[i] = v; cum[i] = v;
    }
    for (size_t i = i0; i < (size_t)BATCH * 1792; i += st) acat[i]  = (_Float16)0.f;
    for (size_t i = i0; i < (size_t)BATCH * 2560; i += st) pacat[i] = (_Float16)0.f;
    for (size_t i = i0; i < (size_t)BATCH * 1536; i += st) pjacat[i] = (_Float16)0.f;
}

// ---------------------------------------------------------------------------
// Persistent decoder
// ---------------------------------------------------------------------------
struct DecArgs {
    const float *enc, *preb, *postb, *projb, *wq, *wconv, *wloc, *av, *pm;
    const int* lens;
    const _Float16 *pwPre, *pwPost, *pwProj, *xf16;
    _Float16 *acat, *pacat, *pjacat;               // f16 activation rows
    float *gpre, *gpost, *preh, *prec, *postc, *ctx, *algn, *cum;
    float *outProj, *outAlign;
    unsigned* bar;
};

__device__ __forceinline__ void proj_store(const DecArgs& A, int tile, int lane, int tOut) {
    v8f c = wmma_tile(A.pjacat, 1536, A.pwProj + (size_t)tile * 48 * 512, 48, lane);
    int col = tile * 16 + (lane & 15), half = lane >> 4;
    if (col < FEATD) {
        float bb = A.projb[col];
#pragma unroll
        for (int r = 0; r < 8; ++r) {
            int b = r + half * 8;
            A.outProj[((size_t)b * FEATD + col) * DEC_T + tOut] = c[r] + bb;
        }
    }
}

__global__ __launch_bounds__(256)
void decoder_kernel(DecArgs A) {
    const int tid = threadIdx.x, bid = blockIdx.x;
    const int wave = bid * 8 + (tid >> 5), lane = tid & 31;
    __shared__ __align__(32) _Float16 s_a[BATCH * 1792];   // TDM-staged pre-LSTM A
    __shared__ float s_al[ENC_TT], s_cu[ENC_TT], s_q[128], s_e[ENC_TT], s_red[ENC_TT];

    // prologue: xt(0) into the pre-LSTM activation row [xt | ctx | pre_h]
    if (bid == 0)
        for (int i = tid; i < BATCH * 256; i += 256) {
            int b = i >> 8, c = i & 255;
            A.acat[(size_t)b * 1792 + c] = A.xf16[(size_t)b * 256 + c];
        }
    grid_sync(A.bar);

    for (int t = 0; t < DEC_T; ++t) {
        // ---- P1: TDM-stage acat to LDS; pre-LSTM gates (256 tiles) + proj(t-1)
        if (tid < 32) tdm_load_1d(A.acat, s_a, BATCH * 1792);  // wave 0 issues DMA + waits
        __syncthreads();
        {
            v8f c = wmma_tile(s_a, 1792, A.pwPre + (size_t)wave * 56 * 512, 56, lane);
            int col = wave * 16 + (lane & 15), half = lane >> 4;
#pragma unroll
            for (int r = 0; r < 8; ++r) A.gpre[(size_t)(r + half * 8) * 4096 + col] = c[r];
            if (t > 0 && wave < 65) proj_store(A, wave, lane, t - 1);
        }
        grid_sync(A.bar);

        // ---- P2: pre-LSTM cell update ------------------------------------
        for (int i = bid * 256 + tid; i < BATCH * 1024; i += NBLK * 256) {
            int b = i >> 10, j = i & 1023;
            const float* g = A.gpre + (size_t)b * 4096;
            float gi = g[j]        + A.preb[j];
            float gf = g[1024 + j] + A.preb[1024 + j];
            float gg = g[2048 + j] + A.preb[2048 + j];
            float gO = g[3072 + j] + A.preb[3072 + j];
            float cc = sigm(gf) * A.prec[i] + sigm(gi) * tanhf(gg);
            float hh = sigm(gO) * tanhf(cc);
            A.prec[i] = cc;
            hh = (rng01(0x3333u, (unsigned)(t * 16384 + i)) < 0.9f) ? hh * (1.0f / 0.9f) : 0.0f;
            A.preh[i] = hh;
            _Float16 h16 = (_Float16)hh;
            A.pacat[(size_t)b * 2560 + j]       = h16;   // post-LSTM input
            A.acat [(size_t)b * 1792 + 768 + j] = h16;   // recurrent input for t+1
        }
        grid_sync(A.bar);

        // ---- P3: attention (blocks 0..15); blocks 16..31 prefetch xt(t+1) --
        if (bid < BATCH) {
            const int b = bid;
            s_al[tid] = A.algn[b * ENC_TT + tid];
            s_cu[tid] = A.cum [b * ENC_TT + tid];
            __syncthreads();
            if (tid < 128) {                          // q = pre_h @ Wq
                float acc = 0.0f;
                const float* ph = A.preh + b * 1024;
                for (int d = 0; d < 1024; ++d) acc += ph[d] * A.wq[d * 128 + tid];
                s_q[tid] = acc;
            }
            __syncthreads();
            {
                const int tt = tid;
                float co[32];
#pragma unroll
                for (int ch = 0; ch < 32; ++ch) co[ch] = 0.0f;
                for (int k = 0; k < 31; ++k) {        // location conv (2->32, k=31)
                    int p = tt + k - 15;
                    if (p >= 0 && p < ENC_TT) {
                        float xa = s_al[p], xc = s_cu[p];
#pragma unroll
                        for (int ch = 0; ch < 32; ++ch)
                            co[ch] += A.wconv[(ch * 2 + 0) * 31 + k] * xa
                                    + A.wconv[(ch * 2 + 1) * 31 + k] * xc;
                    }
                }
                float e = 0.0f;
                const float* pmp = A.pm + ((size_t)b * ENC_TT + tt) * 128;
                for (int j = 0; j < 128; ++j) {
                    float s = s_q[j] + pmp[j];
#pragma unroll
                    for (int ch = 0; ch < 32; ++ch) s += co[ch] * A.wloc[ch * 128 + j];
                    e += A.av[j] * tanhf(s);
                }
                if (tt >= A.lens[b]) e = -1e9f;
                s_e[tid] = e;
            }
            __syncthreads();
            s_red[tid] = s_e[tid]; __syncthreads();
            for (int s = 128; s > 0; s >>= 1) {
                if (tid < s) s_red[tid] = fmaxf(s_red[tid], s_red[tid + s]);
                __syncthreads();
            }
            float mx = s_red[0]; __syncthreads();
            float ex = __expf(s_e[tid] - mx);
            s_red[tid] = ex; __syncthreads();
            for (int s = 128; s > 0; s >>= 1) {
                if (tid < s) s_red[tid] += s_red[tid + s];
                __syncthreads();
            }
            float al = ex / s_red[0];
            s_al[tid] = al;
            A.algn[b * ENC_TT + tid] = al;
            A.cum [b * ENC_TT + tid] = s_cu[tid] + al;
            A.outAlign[((size_t)b * ENC_TT + tid) * DEC_T + t] = al;
            __syncthreads();
            for (int d = tid; d < ENC_DD; d += 256) {     // ctx = align @ mem
                const float* er = A.enc + ((size_t)b * ENC_DD + d) * ENC_TT;
                float acc = 0.0f;
                for (int tt = 0; tt < ENC_TT; ++tt) acc += s_al[tt] * er[tt];
                A.ctx[b * ENC_DD + d] = acc;
                _Float16 h = (_Float16)acc;
                A.acat  [(size_t)b * 1792 + 256  + d] = h;
                A.pacat [(size_t)b * 2560 + 1024 + d] = h;
                A.pjacat[(size_t)b * 1536 + 1024 + d] = h;
            }
        } else {
            for (int i = (bid - BATCH) * 256 + tid; i < BATCH * 256; i += BATCH * 256) {
                int b = i >> 8, c = i & 255;
                A.acat[(size_t)b * 1792 + c] =
                    (t + 1 < DEC_T) ? A.xf16[((size_t)(t + 1) * BATCH + b) * 256 + c]
                                    : (_Float16)0.0f;
            }
        }
        grid_sync(A.bar);

        // ---- P4: post-LSTM gates (256 tiles, K=2560) ----------------------
        {
            v8f c = wmma_tile(A.pacat, 2560, A.pwPost + (size_t)wave * 80 * 512, 80, lane);
            int col = wave * 16 + (lane & 15), half = lane >> 4;
#pragma unroll
            for (int r = 0; r < 8; ++r) A.gpost[(size_t)(r + half * 8) * 4096 + col] = c[r];
        }
        grid_sync(A.bar);

        // ---- P5: post-LSTM cell update -----------------------------------
        for (int i = bid * 256 + tid; i < BATCH * 1024; i += NBLK * 256) {
            int b = i >> 10, j = i & 1023;
            const float* g = A.gpost + (size_t)b * 4096;
            float gi = g[j]        + A.postb[j];
            float gf = g[1024 + j] + A.postb[1024 + j];
            float gg = g[2048 + j] + A.postb[2048 + j];
            float gO = g[3072 + j] + A.postb[3072 + j];
            float cc = sigm(gf) * A.postc[i] + sigm(gi) * tanhf(gg);
            float hh = sigm(gO) * tanhf(cc);
            A.postc[i] = cc;
            hh = (rng01(0x4444u, (unsigned)(t * 16384 + i)) < 0.9f) ? hh * (1.0f / 0.9f) : 0.0f;
            _Float16 h16 = (_Float16)hh;
            A.pjacat[(size_t)b * 1536 + j]        = h16;  // projection input
            A.pacat [(size_t)b * 2560 + 1536 + j] = h16;  // recurrent for t+1
        }
        grid_sync(A.bar);
    }
    // epilogue: projection of the final step
    if (wave < 65) proj_store(A, wave, lane, DEC_T - 1);
}

// ---------------------------------------------------------------------------
// Host launcher
// ---------------------------------------------------------------------------
static inline size_t al256(size_t x) { return (x + 255) & ~(size_t)255; }

extern "C" void kernel_launch(void* const* d_in, const int* in_sizes, int n_in,
                              void* d_out, int out_size, void* d_ws, size_t ws_size,
                              hipStream_t stream) {
    (void)in_sizes; (void)n_in; (void)out_size; (void)ws_size;
    const float* enc     = (const float*)d_in[0];
    const int*   lens    = (const int*)d_in[1];
    const float* feats   = (const float*)d_in[2];
    const float* go      = (const float*)d_in[3];
    const float* w0      = (const float*)d_in[4];
    const float* w1      = (const float*)d_in[5];
    const float* preWih  = (const float*)d_in[6];
    const float* preWhh  = (const float*)d_in[7];
    const float* preB    = (const float*)d_in[8];
    const float* wq      = (const float*)d_in[9];
    const float* wm      = (const float*)d_in[10];
    const float* wconv   = (const float*)d_in[11];
    const float* wloc    = (const float*)d_in[12];
    const float* av      = (const float*)d_in[13];
    const float* postWih = (const float*)d_in[14];
    const float* postWhh = (const float*)d_in[15];
    const float* postB   = (const float*)d_in[16];
    const float* projW   = (const float*)d_in[17];
    const float* projB   = (const float*)d_in[18];

    uint8_t* ws = (uint8_t*)d_ws;
    size_t off = 0;
    auto alloc = [&](size_t bytes) -> void* { void* p = ws + off; off = al256(off + bytes); return p; };

    unsigned*  bar    = (unsigned*)alloc(256);
    _Float16*  pwPre  = (_Float16*)alloc(2ull * 56 * 256 * 512);   // [xt|ctx ; pre_h] stacked
    _Float16*  pwPost = (_Float16*)alloc(2ull * 80 * 256 * 512);   // [pre_h|ctx ; post_h] stacked
    _Float16*  pwProj = (_Float16*)alloc(2ull * 48 * 65  * 512);
    _Float16*  pwP0   = (_Float16*)alloc(2ull * 33 * 16  * 512);
    _Float16*  pwP1   = (_Float16*)alloc(2ull * 8  * 16  * 512);
    _Float16*  f16in  = (_Float16*)alloc(2ull * DEC_T * BATCH * 1056);
    _Float16*  xf16   = (_Float16*)alloc(2ull * DEC_T * BATCH * 256);
    float*     pm     = (float*)alloc(4ull * BATCH * ENC_TT * 128);
    _Float16*  acat   = (_Float16*)alloc(2ull * BATCH * 1792);
    _Float16*  pacat  = (_Float16*)alloc(2ull * BATCH * 2560);
    _Float16*  pjacat = (_Float16*)alloc(2ull * BATCH * 1536);
    float*     gpre   = (float*)alloc(4ull * BATCH * 4096);
    float*     gpost  = (float*)alloc(4ull * BATCH * 4096);
    float*     preh   = (float*)alloc(4ull * BATCH * 1024);
    float*     prec   = (float*)alloc(4ull * BATCH * 1024);
    float*     postc  = (float*)alloc(4ull * BATCH * 1024);
    float*     ctx    = (float*)alloc(4ull * BATCH * ENC_DD);
    float*     algn   = (float*)alloc(4ull * BATCH * ENC_TT);
    float*     cum    = (float*)alloc(4ull * BATCH * ENC_TT);

    float* outProj  = (float*)d_out;                                   // [16,1025,500]
    float* outAlign = outProj + (size_t)BATCH * FEATD * DEC_T;         // [16,256,500]

    init_state<<<64, 256, 0, stream>>>(preh, prec, postc, ctx, algn, cum,
                                       acat, pacat, pjacat, bar);

    pack_weights<<<256 * 24, 32, 0, stream>>>(preWih,  pwPre,  768,  4096, 24, 0,  56);
    pack_weights<<<256 * 32, 32, 0, stream>>>(preWhh,  pwPre,  1024, 4096, 32, 24, 56);
    pack_weights<<<256 * 48, 32, 0, stream>>>(postWih, pwPost, 1536, 4096, 48, 0,  80);
    pack_weights<<<256 * 32, 32, 0, stream>>>(postWhh, pwPost, 1024, 4096, 32, 48, 80);
    pack_weights<<<65  * 48, 32, 0, stream>>>(projW,   pwProj, 1536, 1025, 48, 0,  48);
    pack_weights<<<16  * 33, 32, 0, stream>>>(w0,      pwP0,   1025, 256,  33, 0,  33);
    pack_weights<<<16  * 8,  32, 0, stream>>>(w1,      pwP1,   256,  256,  8,  0,  8);

    build_feats<<<4096, 256, 0, stream>>>(feats, go, f16in);
    prenet_kernel<<<DEC_T, 256, 0, stream>>>(f16in, pwP0, pwP1, xf16);
    proc_mem_kernel<<<BATCH * ENC_TT, 128, 0, stream>>>(enc, wm, pm);

    DecArgs A;
    A.enc = enc; A.preb = preB; A.postb = postB; A.projb = projB;
    A.wq = wq; A.wconv = wconv; A.wloc = wloc; A.av = av; A.pm = pm;
    A.lens = lens;
    A.pwPre = pwPre; A.pwPost = pwPost; A.pwProj = pwProj; A.xf16 = xf16;
    A.acat = acat; A.pacat = pacat; A.pjacat = pjacat;
    A.gpre = gpre; A.gpost = gpost; A.preh = preh; A.prec = prec; A.postc = postc;
    A.ctx = ctx; A.algn = algn; A.cum = cum;
    A.outProj = outProj; A.outAlign = outAlign; A.bar = bar;

    decoder_kernel<<<NBLK, 256, 0, stream>>>(A);
}
